// Attention_90984587199026
// MI455X (gfx1250) — matrix operand order repaired and checked by reading, not compiled, not run on hardware
//
#include <hip/hip_runtime.h>
#include <hip/hip_bf16.h>

// ---------------------------------------------------------------------------
// Types for CDNA5 WMMA (wave32): v16 bf16 A/B fragments, v8 f32 accumulator.
// ---------------------------------------------------------------------------
typedef __bf16 bf16_t;
typedef __attribute__((ext_vector_type(16))) __bf16 v16bf;
typedef __attribute__((ext_vector_type(8)))  __bf16 v8bf;
typedef __attribute__((ext_vector_type(8)))  float  v8f;
typedef __attribute__((ext_vector_type(4)))  unsigned int u32x4;
typedef __attribute__((ext_vector_type(8)))  unsigned int u32x8;

// Problem constants (from the reference)
constexpr int BATCH  = 2;
constexpr int NSEQ   = 2048;
constexpr int DIM    = 1020;
constexpr int HEADS  = 17;
constexpr int DH     = 60;            // head dim
constexpr int ROWS   = BATCH * NSEQ;  // 4096 flattened (b, n) rows
constexpr int KPAD   = 1024;          // contraction dim padded (1020 -> 1024)
constexpr int NQKV_PAD = 3072;        // 3060 -> 3072 (multiple of 128)
constexpr int BH     = BATCH * HEADS; // 34
constexpr float QSCALE = 0.12909944487358056f; // 60^-0.5

// GEMM LDS tile geometry: 128 rows x 16 dwords (32 bf16), TDM-padded by
// 4 dwords per row -> LDS row stride 20 dwords = 40 bf16 = 80 B (16B aligned,
// bank-spread). One tile = 128*80 = 10240 B.
constexpr int LDS_LD     = 40;     // bf16 elements per LDS row
constexpr int TILE_BYTES = 128 * LDS_LD * 2;

// ---------------------------------------------------------------------------
// WMMA helpers
// ---------------------------------------------------------------------------
__device__ inline v8f wmma_bf16(v16bf a, v16bf b, v8f c) {
  // D = A(16x32 bf16) * B(32x16 bf16) + C(16x16 f32)
  return __builtin_amdgcn_wmma_f32_16x16x32_bf16(
      /*neg_a=*/false, a, /*neg_b=*/false, b,
      /*c_mod=*/(short)0, c, /*reuse_a=*/false, /*reuse_b=*/false);
}

// A-fragment: source row-major [rows][ld]; rows row0..row0+15, cols k0..k0+31.
// ISA layout: lane<16 holds K {0..7,16..23}, lane>=16 holds K {8..15,24..31}.
__device__ inline v16bf load_frag_a(const bf16_t* base, int ld, int row0,
                                    int k0, int lane) {
  int r  = row0 + (lane & 15);
  int ko = k0 + ((lane & 16) ? 8 : 0);
  const bf16_t* p = base + (size_t)r * ld + ko;
  v8bf lo = *(const v8bf*)p;
  v8bf hi = *(const v8bf*)(p + 16);
  v16bf f;
#pragma unroll
  for (int i = 0; i < 8; ++i) { f[i] = lo[i]; f[i + 8] = hi[i]; }
  return f;
}

// B-fragment: source stored K-transposed as [N][ld]. lane<16 -> K {0..7,16..23},
// lane>=16 -> K {8..15,24..31}.
__device__ inline v16bf load_frag_b(const bf16_t* base, int ld, int n0,
                                    int k0, int lane) {
  int c  = n0 + (lane & 15);
  int ko = k0 + ((lane & 16) ? 8 : 0);
  const bf16_t* p = base + (size_t)c * ld + ko;
  v8bf lo = *(const v8bf*)p;
  v8bf hi = *(const v8bf*)(p + 16);
  v16bf f;
#pragma unroll
  for (int i = 0; i < 8; ++i) { f[i] = lo[i]; f[i + 8] = hi[i]; }
  return f;
}

// ---------------------------------------------------------------------------
// 16-lane butterfly reductions via DPP (pure VALU, no LDS waits).
// 0xB1 = quad_perm(1,0,3,2) (xor1), 0x4E = quad_perm(2,3,0,1) (xor2),
// 0x141 = row_half_mirror (pairs quads), 0x140 = row_mirror (pairs octets).
// ---------------------------------------------------------------------------
#if __has_builtin(__builtin_amdgcn_update_dpp)
template <int CTRL>
__device__ inline float dpp_xfer(float x) {
  int r = __builtin_amdgcn_update_dpp(0, __builtin_bit_cast(int, x),
                                      CTRL, 0xF, 0xF, true);
  return __builtin_bit_cast(float, r);
}
__device__ inline float red_max16(float x) {
  x = fmaxf(x, dpp_xfer<0xB1>(x));
  x = fmaxf(x, dpp_xfer<0x4E>(x));
  x = fmaxf(x, dpp_xfer<0x141>(x));
  x = fmaxf(x, dpp_xfer<0x140>(x));
  return x;
}
__device__ inline float red_sum16(float x) {
  x += dpp_xfer<0xB1>(x);
  x += dpp_xfer<0x4E>(x);
  x += dpp_xfer<0x141>(x);
  x += dpp_xfer<0x140>(x);
  return x;
}
#else
__device__ inline float red_max16(float x) {
  x = fmaxf(x, __shfl_xor(x, 1)); x = fmaxf(x, __shfl_xor(x, 2));
  x = fmaxf(x, __shfl_xor(x, 4)); x = fmaxf(x, __shfl_xor(x, 8));
  return x;
}
__device__ inline float red_sum16(float x) {
  x += __shfl_xor(x, 1); x += __shfl_xor(x, 2);
  x += __shfl_xor(x, 4); x += __shfl_xor(x, 8);
  return x;
}
#endif

// ---------------------------------------------------------------------------
// Tensor Data Mover: async 2D tile load (128 rows x 16 dwords) into LDS.
// D# built per CDNA5 ISA ch.8: group0 = {count, lds_addr, global_addr, type=2},
// group1 = {data_size=4B, pad 4dw per 16dw, dims/strides}. Issued from one
// wave; completion tracked with TENSORcnt.
// ---------------------------------------------------------------------------
__device__ inline void tdm_load_tile_128x16dw(unsigned lds_off,
                                              const bf16_t* gptr,
                                              unsigned row_stride_dw) {
  unsigned long long ga = (unsigned long long)(uintptr_t)gptr;
  u32x4 g0;
  g0[0] = 1u;                                     // count=1 (user descriptor)
  g0[1] = lds_off;                                // lds_addr (bytes)
  g0[2] = (unsigned)ga;                           // global_addr[31:0]
  g0[3] = (unsigned)((ga >> 32) & 0x01ffffffu)    // global_addr[56:32]
          | (2u << 30);                           // type = 2 ("image")
  u32x8 g1;
  g1[0] = (2u << 16)      // data_size = 4 bytes
        | (1u << 20)      // pad_enable
        | (3u << 22)      // pad_interval: 16 dwords
        | (3u << 25);     // pad_amount: 4 dwords
  g1[1] = (16u << 16);    // tensor_dim0 = 16 dw (tile-local, no OOB)
  g1[2] = (128u << 16);   // tensor_dim0 hi | tensor_dim1 = 128
  g1[3] = (16u << 16);    // tensor_dim1 hi | tile_dim0 = 16 dw
  g1[4] = 128u;           // tile_dim1 = 128, tile_dim2 = 0
  g1[5] = row_stride_dw;  // tensor_dim0_stride[31:0]
  g1[6] = 0u;             // stride hi | tensor_dim1_stride lo
  g1[7] = 0u;
  asm volatile("tensor_load_to_lds %0, %1" :: "s"(g0), "s"(g1) : "memory");
}

// ---------------------------------------------------------------------------
// fp32 -> bf16 convert with zero padding (row-major copy)
// ---------------------------------------------------------------------------
__global__ void convert_pad_kernel(const float* __restrict__ src,
                                   bf16_t* __restrict__ dst,
                                   int sr, int sc, int dr, int dc) {
  int idx = blockIdx.x * blockDim.x + threadIdx.x;
  if (idx >= dr * dc) return;
  int r = idx / dc, c = idx - r * dc;
  float v = (r < sr && c < sc) ? src[(size_t)r * sc + c] : 0.f;
  dst[idx] = (bf16_t)v;
}

// fp32 -> bf16 convert with transpose + zero padding: dst[n][k] = src[k][n]
__global__ void convert_pad_t_kernel(const float* __restrict__ src,
                                     bf16_t* __restrict__ dst,
                                     int sr, int sc, int dr, int dc) {
  int idx = blockIdx.x * blockDim.x + threadIdx.x;
  if (idx >= dr * dc) return;
  int r = idx / dc, c = idx - r * dc;          // r = n (out col), c = k
  float v = (c < sr && r < sc) ? src[(size_t)c * sc + r] : 0.f;
  dst[idx] = (bf16_t)v;
}

// Zero the K-pad columns (1020..1023) of the attention-output staging buffer.
__global__ void zero_pad_kernel(bf16_t* __restrict__ attnb) {
  int i = blockIdx.x * blockDim.x + threadIdx.x;
  if (i >= ROWS * 4) return;
  int r = i >> 2, c = i & 3;
  attnb[(size_t)r * KPAD + DIM + c] = (bf16_t)0.f;
}

// ---------------------------------------------------------------------------
// WMMA GEMM: C[M x Nout] = A[M x K](bf16) * Bt[Npad x K](bf16, K-transposed).
// 256 threads = 8 waves; 128x128 C tile; each wave owns 2x4 16x16 tiles.
// Both operand tiles are DMA'd into LDS by the Tensor Data Mover, double
// buffered so the TDM fetch of tile k+1 overlaps the WMMAs of tile k.
// Dynamic LDS: 4 * TILE_BYTES = 40960 B.
// ---------------------------------------------------------------------------
template <typename OutT, bool BIAS>
__global__ __launch_bounds__(256) void gemm_wmma_kernel(
    const bf16_t* __restrict__ A, int lda,
    const bf16_t* __restrict__ Bt, int ldbt,
    OutT* __restrict__ C, int ldc, int Nout,
    const float* __restrict__ bias, int K) {
  extern __shared__ __align__(16) char smem[];

  const int tid  = threadIdx.x;
  const int wave = tid >> 5;
  const int lane = tid & 31;
  const int wm   = wave & 3;   // 0..3 -> 32-row strip
  const int wn   = wave >> 2;  // 0..1 -> 64-col strip
  const int m0   = blockIdx.y * 128;
  const int n0   = blockIdx.x * 128;

  v8f acc[2][4];
#pragma unroll
  for (int mt = 0; mt < 2; ++mt)
#pragma unroll
    for (int nt = 0; nt < 4; ++nt)
      acc[mt][nt] = v8f{0.f, 0.f, 0.f, 0.f, 0.f, 0.f, 0.f, 0.f};

  // Prologue: DMA the k0 = 0 tiles into buffer 0.
  if (wave == 0) {
    tdm_load_tile_128x16dw(0, A + (size_t)m0 * lda, (unsigned)(lda >> 1));
    tdm_load_tile_128x16dw(TILE_BYTES, Bt + (size_t)n0 * ldbt,
                           (unsigned)(ldbt >> 1));
    __builtin_amdgcn_s_wait_tensorcnt(0);
  }
  __syncthreads();

  int buf = 0;
  for (int k0 = 0; k0 < K; k0 += 32) {
    const bool more = (k0 + 32) < K;
    // Kick off the DMA for the next tiles into the other buffer.
    if (wave == 0 && more) {
      unsigned nb = (unsigned)(buf ^ 1) * 2u * TILE_BYTES;
      tdm_load_tile_128x16dw(nb, A + (size_t)m0 * lda + (k0 + 32),
                             (unsigned)(lda >> 1));
      tdm_load_tile_128x16dw(nb + TILE_BYTES,
                             Bt + (size_t)n0 * ldbt + (k0 + 32),
                             (unsigned)(ldbt >> 1));
    }
    const bf16_t* As = (const bf16_t*)(smem + buf * 2 * TILE_BYTES);
    const bf16_t* Bs = (const bf16_t*)(smem + buf * 2 * TILE_BYTES + TILE_BYTES);

    v16bf af[2];
#pragma unroll
    for (int mt = 0; mt < 2; ++mt)
      af[mt] = load_frag_a(As, LDS_LD, wm * 32 + mt * 16, 0, lane);
    v16bf bfv[4];
#pragma unroll
    for (int nt = 0; nt < 4; ++nt)
      bfv[nt] = load_frag_b(Bs, LDS_LD, wn * 64 + nt * 16, 0, lane);
#pragma unroll
    for (int mt = 0; mt < 2; ++mt)
#pragma unroll
      for (int nt = 0; nt < 4; ++nt)
        acc[mt][nt] = wmma_bf16(af[mt], bfv[nt], acc[mt][nt]);

    if (wave == 0 && more) __builtin_amdgcn_s_wait_tensorcnt(0);
    __syncthreads();
    buf ^= 1;
  }

  // Store: C layout — VGPR v, lane L: M = v + 8*(L/16), N = L%16.
  const int mlane = (lane >> 4) << 3;
  const int nlane = lane & 15;
#pragma unroll
  for (int mt = 0; mt < 2; ++mt) {
#pragma unroll
    for (int nt = 0; nt < 4; ++nt) {
      int col = n0 + wn * 64 + nt * 16 + nlane;
      if (col >= Nout) continue;
      float bv = BIAS ? bias[col] : 0.f;
#pragma unroll
      for (int v = 0; v < 8; ++v) {
        int row = m0 + wm * 32 + mt * 16 + mlane + v;
        C[(size_t)row * ldc + col] = (OutT)(acc[mt][nt][v] + bv);
      }
    }
  }
}

// ---------------------------------------------------------------------------
// 3D RoPE + head split.  qkv (bf16 [ROWS][NQKV_PAD]) -> Qh/Kh/Vh
// (bf16 [BH][NSEQ][64], d padded 60->64 with zeros, Q pre-scaled by DH^-0.5)
// ---------------------------------------------------------------------------
__global__ void rope_split_kernel(const bf16_t* __restrict__ qkv,
                                  const float* __restrict__ coords,
                                  bf16_t* __restrict__ Qh,
                                  bf16_t* __restrict__ Kh,
                                  bf16_t* __restrict__ Vh) {
  int idx = blockIdx.x * blockDim.x + threadIdx.x;
  const int total = ROWS * HEADS * 64;
  if (idx >= total) return;
  int dd  = idx & 63;
  int t   = idx >> 6;
  int h   = t % HEADS;
  int row = t / HEADS;                 // flattened (b, n)
  int b   = row >> 11;
  int nl  = row & (NSEQ - 1);
  size_t obase = (((size_t)(b * HEADS + h)) * NSEQ + nl) * 64 + dd;
  if (dd >= DH) {
    Qh[obase] = (bf16_t)0.f; Kh[obase] = (bf16_t)0.f; Vh[obase] = (bf16_t)0.f;
    return;
  }
  const bf16_t* qrow = qkv + (size_t)row * NQKV_PAD;
  int col = h * DH + dd;
  float qv = (float)qrow[col];
  float kv = (float)qrow[DIM + col];
  float vv = (float)qrow[2 * DIM + col];

  int axis = dd / 20;                  // which coordinate (x, y, z)
  int dp   = dd % 20;                  // position inside the 20-wide chunk
  int i    = dp % 10;                  // frequency index
  int partner = (dp < 10) ? dd + 10 : dd - 10;
  float sgn   = (dp < 10) ? -1.f : 1.f;
  float qp = (float)qrow[h * DH + partner] * sgn;
  float kp = (float)qrow[DIM + h * DH + partner] * sgn;

  float coord = coords[(size_t)row * 3 + axis];
  // inv_freq[i] = 10000^(-i/10);  f = (coord / MIN_FREQ) * inv_freq
  float invf = __expf(-(float)i * 0.1f * 9.210340371976184f);
  float f    = coord * 64.0f * invf;
  float cs   = __cosf(f);
  float sn   = __sinf(f);

  Qh[obase] = (bf16_t)((qv * cs + qp * sn) * QSCALE);
  Kh[obase] = (bf16_t)(kv * cs + kp * sn);
  Vh[obase] = (bf16_t)vv;
}

// ---------------------------------------------------------------------------
// Flash attention: grid (NSEQ/64, BH); 128 threads = 4 waves, 16 q-rows/wave.
// Scores and P*V via v_wmma_f32_16x16x32_bf16; online softmax in fp32 with
// DPP butterfly reductions.
// ---------------------------------------------------------------------------
__global__ __launch_bounds__(128) void attn_kernel(
    const bf16_t* __restrict__ Qh, const bf16_t* __restrict__ Kh,
    const bf16_t* __restrict__ Vh, bf16_t* __restrict__ attnb) {
  __shared__ __align__(16) bf16_t Vt[64][72];        // V^T tile: [d][key]
  __shared__ __align__(16) bf16_t Pl[4][16][72];     // per-wave P scratch

  const int bh   = blockIdx.y;
  const int b    = bh / HEADS;
  const int h    = bh - b * HEADS;
  const int wave = threadIdx.x >> 5;
  const int lane = threadIdx.x & 31;
  const int r0   = blockIdx.x * 64 + wave * 16;      // first q row (this wave)

  const bf16_t* Q  = Qh + (size_t)bh * NSEQ * 64;
  const bf16_t* Kg = Kh + (size_t)bh * NSEQ * 64;
  const bf16_t* Vg = Vh + (size_t)bh * NSEQ * 64;

  // Q fragments (held for the whole kernel): rows r0..r0+15, d 0..63.
  v16bf aq0 = load_frag_a(Q, 64, r0, 0, lane);
  v16bf aq1 = load_frag_a(Q, 64, r0, 32, lane);

  float m[8], l[8];
  v8f acc[4];
#pragma unroll
  for (int v = 0; v < 8; ++v) { m[v] = -1e30f; l[v] = 0.f; }
#pragma unroll
  for (int dt = 0; dt < 4; ++dt)
    acc[dt] = v8f{0.f, 0.f, 0.f, 0.f, 0.f, 0.f, 0.f, 0.f};

  for (int j0 = 0; j0 < NSEQ; j0 += 64) {
    __syncthreads();  // previous iteration finished consuming Vt
    // Stage V^T: 64 keys x 64 d -> Vt[d][key]; 512 8-elt chunks, 4/thread.
#pragma unroll
    for (int i = 0; i < 4; ++i) {
      int c  = threadIdx.x + i * 128;
      int k  = c >> 3;
      int d8 = (c & 7) * 8;
      v8bf v = *(const v8bf*)(Vg + (size_t)(j0 + k) * 64 + d8);
#pragma unroll
      for (int j = 0; j < 8; ++j) Vt[d8 + j][k] = v[j];
    }
    if (j0 + 64 < NSEQ) {  // hint next tiles toward the caches
      __builtin_prefetch(Kg + (size_t)(j0 + 64 + (threadIdx.x >> 1)) * 64, 0, 1);
      __builtin_prefetch(Vg + (size_t)(j0 + 64 + (threadIdx.x >> 1)) * 64, 0, 1);
    }

    // Scores: S = Q * K^T. K rows ([key][d]) are exactly B-fragment layout.
    v8f s[4];
#pragma unroll
    for (int jt = 0; jt < 4; ++jt) {
      v16bf bk0 = load_frag_b(Kg, 64, j0 + jt * 16, 0, lane);
      v16bf bk1 = load_frag_b(Kg, 64, j0 + jt * 16, 32, lane);
      v8f z = v8f{0.f, 0.f, 0.f, 0.f, 0.f, 0.f, 0.f, 0.f};
      z = wmma_bf16(aq0, bk0, z);
      z = wmma_bf16(aq1, bk1, z);
      s[jt] = z;
    }

    // Online softmax update (rows replicated across 16-lane groups).
#pragma unroll
    for (int v = 0; v < 8; ++v) {
      float mx = fmaxf(fmaxf(s[0][v], s[1][v]), fmaxf(s[2][v], s[3][v]));
      mx = red_max16(mx);
      float mnew = fmaxf(m[v], mx);
      float corr = __expf(m[v] - mnew);
      m[v] = mnew;
      float ps = 0.f;
#pragma unroll
      for (int jt = 0; jt < 4; ++jt) {
        float p = __expf(s[jt][v] - mnew);
        s[jt][v] = p;
        ps += p;
      }
      ps = red_sum16(ps);
      l[v] = l[v] * corr + ps;
#pragma unroll
      for (int dt = 0; dt < 4; ++dt) acc[dt][v] *= corr;
    }

    // P (C-layout) -> LDS -> A-fragments for P*V.
    const int prow = (lane >> 4) << 3;
#pragma unroll
    for (int jt = 0; jt < 4; ++jt)
#pragma unroll
      for (int v = 0; v < 8; ++v)
        Pl[wave][prow + v][jt * 16 + (lane & 15)] = (bf16_t)s[jt][v];
    asm volatile("s_wait_dscnt 0" ::: "memory");  // per-wave LDS RAW
    v16bf ap0 = load_frag_a(&Pl[wave][0][0], 72, 0, 0, lane);
    v16bf ap1 = load_frag_a(&Pl[wave][0][0], 72, 0, 32, lane);

    __syncthreads();  // Vt staging complete across all waves
#pragma unroll
    for (int dt = 0; dt < 4; ++dt) {
      v16bf bv0 = load_frag_b(&Vt[0][0], 72, dt * 16, 0, lane);
      v16bf bv1 = load_frag_b(&Vt[0][0], 72, dt * 16, 32, lane);
      acc[dt] = wmma_bf16(ap0, bv0, acc[dt]);
      acc[dt] = wmma_bf16(ap1, bv1, acc[dt]);
    }
  }

  // Normalize and scatter to [row][h*60 + d] staging buffer (bf16).
  const int mlane = (lane >> 4) << 3;
  const int nlane = lane & 15;
#pragma unroll
  for (int v = 0; v < 8; ++v) {
    float inv = 1.f / l[v];
    int row = r0 + mlane + v;
#pragma unroll
    for (int dt = 0; dt < 4; ++dt) {
      int d = dt * 16 + nlane;
      if (d < DH)
        attnb[(size_t)(b * NSEQ + row) * KPAD + h * DH + d] =
            (bf16_t)(acc[dt][v] * inv);
    }
  }
}

// ---------------------------------------------------------------------------
// Host-side orchestration
// ---------------------------------------------------------------------------
extern "C" void kernel_launch(void* const* d_in, const int* in_sizes, int n_in,
                              void* d_out, int out_size, void* d_ws,
                              size_t ws_size, hipStream_t stream) {
  (void)in_sizes; (void)n_in; (void)out_size; (void)ws_size;
  const float* x      = (const float*)d_in[0];
  const float* coords = (const float*)d_in[1];
  const float* w_qkv  = (const float*)d_in[2];
  const float* w_out  = (const float*)d_in[3];
  const float* b_out  = (const float*)d_in[4];
  float* out = (float*)d_out;

  // Workspace layout (bf16 elements)
  bf16_t* ws = (bf16_t*)d_ws;
  size_t o = 0;
  bf16_t* xb    = ws + o; o += (size_t)ROWS * KPAD;        // x, padded
  bf16_t* wqkvT = ws + o; o += (size_t)NQKV_PAD * KPAD;    // w_qkv^T, padded
  bf16_t* woutT = ws + o; o += (size_t)KPAD * KPAD;        // w_out^T, padded
  bf16_t* qkvb  = ws + o; o += (size_t)ROWS * NQKV_PAD;    // qkv result
  bf16_t* Qh    = ws + o; o += (size_t)BH * NSEQ * 64;
  bf16_t* Khd   = ws + o; o += (size_t)BH * NSEQ * 64;
  bf16_t* Vhd   = ws + o; o += (size_t)BH * NSEQ * 64;
  bf16_t* attnb = ws + o; o += (size_t)ROWS * KPAD;        // attn out staging

  const int T = 256;
  const int GEMM_LDS = 4 * TILE_BYTES;  // double-buffered A + B tiles

  // 1) Converts with zero padding (weights transposed for TDM tile loads)
  convert_pad_kernel<<<(ROWS * KPAD + T - 1) / T, T, 0, stream>>>(
      x, xb, ROWS, DIM, ROWS, KPAD);
  convert_pad_t_kernel<<<(NQKV_PAD * KPAD + T - 1) / T, T, 0, stream>>>(
      w_qkv, wqkvT, DIM, 3 * DIM, NQKV_PAD, KPAD);
  convert_pad_t_kernel<<<(KPAD * KPAD + T - 1) / T, T, 0, stream>>>(
      w_out, woutT, DIM, DIM, KPAD, KPAD);
  zero_pad_kernel<<<(ROWS * 4 + T - 1) / T, T, 0, stream>>>(attnb);

  // 2) QKV projection: [4096 x 1024] * [1024 x 3072] -> bf16 qkv
  gemm_wmma_kernel<bf16_t, false>
      <<<dim3(NQKV_PAD / 128, ROWS / 128), 256, GEMM_LDS, stream>>>(
          xb, KPAD, wqkvT, KPAD, qkvb, NQKV_PAD, NQKV_PAD, nullptr, KPAD);

  // 3) 3D RoPE + head split (+ pre-scale Q)
  rope_split_kernel<<<(ROWS * HEADS * 64 + T - 1) / T, T, 0, stream>>>(
      qkvb, coords, Qh, Khd, Vhd);

  // 4) Flash attention per (b, h)
  attn_kernel<<<dim3(NSEQ / 64, BH), 128, 0, stream>>>(Qh, Khd, Vhd, attnb);

  // 5) Output projection + bias: [4096 x 1024] * [1024 x 1024] -> f32 out
  gemm_wmma_kernel<float, true>
      <<<dim3(KPAD / 128, ROWS / 128), 256, GEMM_LDS, stream>>>(
          attnb, KPAD, woutT, KPAD, out, DIM, DIM, b_out, KPAD);
}
